// Qwen35PromptAligner_65403761984400
// MI455X (gfx1250) — matrix operand order repaired and checked by reading, not compiled, not run on hardware
//
#include <hip/hip_runtime.h>

typedef __bf16 bf16_t;
typedef __attribute__((ext_vector_type(16))) __bf16 v16bf;
typedef __attribute__((ext_vector_type(8)))  float  v8f;
typedef __attribute__((ext_vector_type(4)))  unsigned v4u;

#define NUM_GROUPS 12
#define IN_DIM     3072
#define OUT_DIM    3072
#define RANK       1024
#define IN_PG      256
#define OUT_PG     256
#define T_TOK      16           // tokens per workgroup
#define KR         64           // rank chunk
#define NCHUNK     (RANK / KR)  // 16
#define THREADS    512          // 16 waves (wave32)
#define XP         3080         // x_s row pitch (bf16), 16B-aligned rows
#define ZP         776          // z_s row pitch (bf16) = 12*64 + 8, 16B-aligned

union FragCast { struct { v4u lo, hi; } p; v16bf v; };

__device__ __forceinline__ v16bf make_frag(v4u lo, v4u hi) {
  FragCast f; f.p.lo = lo; f.p.hi = hi; return f.v;
}

// ---------------------------------------------------------------------------
// Prep: fp32 weights -> transposed bf16 weights in workspace.
//   down[g][i][r] -> down_bt[g][r][i]   (B operand for down GEMM: N=r, K=i)
//   up[h][r][o]   -> up_bt[h][o][r]     (B operand for up GEMM:   N=o, K=r)
// ---------------------------------------------------------------------------
__global__ __launch_bounds__(256) void prep_kernel(const float* __restrict__ down,
                                                   const float* __restrict__ up,
                                                   bf16_t* __restrict__ down_bt,
                                                   bf16_t* __restrict__ up_bt) {
  long idx = (long)blockIdx.x * 256 + threadIdx.x;   // 0 .. 12*256*1024-1
  // down transpose: out index (g, r, i)
  {
    int i = (int)(idx & 255);
    int r = (int)((idx >> 8) & 1023);
    int g = (int)(idx >> 18);
    down_bt[idx] = (bf16_t)down[((long)g * IN_PG + i) * RANK + r];
  }
  // up transpose: out index (h, o, r)
  {
    int r = (int)(idx & 1023);
    int o = (int)((idx >> 10) & 255);
    int h = (int)(idx >> 18);
    up_bt[idx] = (bf16_t)up[((long)h * RANK + r) * OUT_PG + o];
  }
}

// ---------------------------------------------------------------------------
// Fused: y = ((x_g · down_g) mixed over groups per-rank) · up_h
// One WG = 16 tokens. Loop over rank chunks; z chunk lives in LDS (bf16),
// y accumulates in fp32 registers (12 v8f tiles per lane).
// ---------------------------------------------------------------------------
__global__ __launch_bounds__(THREADS) void fused_kernel(const float* __restrict__ x,
                                                        const bf16_t* __restrict__ down_bt,
                                                        const float* __restrict__ mix,
                                                        const bf16_t* __restrict__ up_bt,
                                                        float* __restrict__ y) {
  __shared__ __align__(16) bf16_t x_s[T_TOK * XP];          // 98,560 B
  __shared__ __align__(16) bf16_t z_s[T_TOK * ZP];          // 24,832 B
  __shared__ __align__(16) bf16_t mix_s[NUM_GROUPS * NUM_GROUPS * KR]; // 18,432 B

  const int tid   = threadIdx.x;
  const int wave  = tid >> 5;        // 0..15
  const int lane  = tid & 31;
  const int lhalf = lane >> 4;       // 0: lanes 0-15, 1: lanes 16-31
  const int l15   = lane & 15;
  const long tok0 = (long)blockIdx.x * T_TOK;

  // ---- stage x tile: fp32 -> bf16 into LDS (coalesced float4 reads) ----
  {
    const float4* xg = (const float4*)(x + tok0 * IN_DIM);
    #pragma unroll
    for (int v = 0; v < 24; ++v) {
      int fi  = tid + v * THREADS;        // 0..12287 float4s
      int row = fi / (IN_DIM / 4);
      int c4  = fi % (IN_DIM / 4);
      float4 f = xg[row * (IN_DIM / 4) + c4];
      int o = row * XP + c4 * 4;
      x_s[o + 0] = (bf16_t)f.x; x_s[o + 1] = (bf16_t)f.y;
      x_s[o + 2] = (bf16_t)f.z; x_s[o + 3] = (bf16_t)f.w;
    }
  }

  v8f acc[NUM_GROUPS];
  #pragma unroll
  for (int h = 0; h < NUM_GROUPS; ++h) acc[h] = {};

  __syncthreads();

  for (int c = 0; c < NCHUNK; ++c) {
    const int r0 = c * KR;
    if (c > 0) __syncthreads();   // WAR: prev phase-3 z reads / phase-2 mix reads done

    // ---- stage mix chunk (fp32 -> bf16): layout [g*12+h][KR] ----
    #pragma unroll
    for (int v = 0; v < 18; ++v) {
      int idx = tid + v * THREADS;        // 0..9215
      int r   = idx & (KR - 1);
      int gh  = idx >> 6;                 // g*12+h
      mix_s[idx] = (bf16_t)mix[(long)gh * RANK + r0 + r];
    }

    // ---- phase 1: z[m, g*64 + r] = x[m, g,:] . down[g,:,r0+r] ----
    // 48 tiles (g 0..11, nb 0..3); wave does tiles {w, w+16, w+32}
    for (int tt = 0; tt < 3; ++tt) {
      int tileid = wave + tt * 16;
      int g  = tileid >> 2;
      int nb = tileid & 3;
      const bf16_t* arow = x_s + l15 * XP + g * IN_PG;
      const bf16_t* brow = down_bt + ((long)g * RANK + (r0 + nb * 16 + l15)) * IN_PG;
      v8f cz = {};
      #pragma unroll
      for (int kb = 0; kb < 8; ++kb) {
        int k0 = kb * 32;
        // A (16x32 bf16): lanes 0-15 K{0-7,16-23}, lanes 16-31 K{8-15,24-31}
        v4u alo = *(const v4u*)(arow + k0 + lhalf * 8);
        v4u ahi = *(const v4u*)(arow + k0 + 16 + lhalf * 8);
        // B (32x16 bf16): lanes 0-15 K{0-7,16-23}, lanes 16-31 K{8-15,24-31}
        v4u blo = *(const v4u*)(brow + k0 + lhalf * 8);
        v4u bhi = *(const v4u*)(brow + k0 + 16 + lhalf * 8);
        cz = __builtin_amdgcn_wmma_f32_16x16x32_bf16(
            false, make_frag(alo, ahi), false, make_frag(blo, bhi),
            (short)0, cz, false, false);
      }
      int colz = g * KR + nb * 16 + l15;
      #pragma unroll
      for (int j = 0; j < 8; ++j)
        z_s[(j + 8 * lhalf) * ZP + colz] = (bf16_t)cz[j];
    }
    __syncthreads();

    // ---- phase 2: per-rank 12x12 group mix, in place in z_s ----
    #pragma unroll
    for (int v = 0; v < 2; ++v) {
      int idx = tid + v * THREADS;        // 0..1023 -> (m, r)
      int m = idx >> 6;
      int r = idx & (KR - 1);
      float zg[NUM_GROUPS];
      #pragma unroll
      for (int g = 0; g < NUM_GROUPS; ++g)
        zg[g] = (float)z_s[m * ZP + g * KR + r];
      float zh[NUM_GROUPS];
      #pragma unroll
      for (int h = 0; h < NUM_GROUPS; ++h) {
        float s = 0.f;
        #pragma unroll
        for (int g = 0; g < NUM_GROUPS; ++g)
          s += zg[g] * (float)mix_s[(g * NUM_GROUPS + h) * KR + r];
        zh[h] = s;
      }
      #pragma unroll
      for (int h = 0; h < NUM_GROUPS; ++h)
        z_s[m * ZP + h * KR + r] = (bf16_t)zh[h];
    }
    __syncthreads();

    // ---- phase 3: y[m, h*256 + wave*16 + n] += z2[m, h,:] . up[h,:,o] ----
    #pragma unroll
    for (int h = 0; h < NUM_GROUPS; ++h) {
      const bf16_t* arow = z_s + l15 * ZP + h * KR;
      const bf16_t* brow = up_bt + ((long)h * OUT_PG + wave * 16 + l15) * RANK + r0;
      v8f ca = acc[h];
      #pragma unroll
      for (int kb = 0; kb < 2; ++kb) {
        int k0 = kb * 32;
        v4u alo = *(const v4u*)(arow + k0 + lhalf * 8);
        v4u ahi = *(const v4u*)(arow + k0 + 16 + lhalf * 8);
        v4u blo = *(const v4u*)(brow + k0 + lhalf * 8);
        v4u bhi = *(const v4u*)(brow + k0 + 16 + lhalf * 8);
        ca = __builtin_amdgcn_wmma_f32_16x16x32_bf16(
            false, make_frag(alo, ahi), false, make_frag(blo, bhi),
            (short)0, ca, false, false);
      }
      acc[h] = ca;
    }
  }

  // ---- write y tile (fp32) ----
  float* yb = y + tok0 * OUT_DIM;
  #pragma unroll
  for (int h = 0; h < NUM_GROUPS; ++h) {
    int col = h * OUT_PG + wave * 16 + l15;
    #pragma unroll
    for (int j = 0; j < 8; ++j)
      yb[(long)(j + 8 * lhalf) * OUT_DIM + col] = acc[h][j];
  }
}

// ---------------------------------------------------------------------------
extern "C" void kernel_launch(void* const* d_in, const int* in_sizes, int n_in,
                              void* d_out, int out_size, void* d_ws, size_t ws_size,
                              hipStream_t stream) {
  const float* x    = (const float*)d_in[0];
  const float* down = (const float*)d_in[1];
  const float* mixw = (const float*)d_in[2];
  const float* up   = (const float*)d_in[3];
  float* y = (float*)d_out;

  const size_t wbytes = (size_t)NUM_GROUPS * RANK * IN_PG * sizeof(bf16_t); // 6,291,456
  bf16_t* down_bt = (bf16_t*)d_ws;
  bf16_t* up_bt   = (bf16_t*)((char*)d_ws + wbytes);

  // 12*256*1024 = 3,145,728 elements; one per thread
  prep_kernel<<<12288, 256, 0, stream>>>(down, up, down_bt, up_bt);

  // 8192 tokens / 16 per WG = 512 workgroups
  fused_kernel<<<512, THREADS, 0, stream>>>(x, down_bt, mixw, up_bt, y);
}
